// ThresholdPredictionGNN_62070867362010
// MI455X (gfx1250) — matrix operand order repaired and checked by reading, not compiled, not run on hardware
//
#include <hip/hip_runtime.h>
#include <hip/hip_bf16.h>

#define NN 50000
#define EE 800000
#define BB 128
#define HH 64
#define LL 4
#define CC 9
#define NGT 20
#define GEW 16
#define MSG_K 96   // 88 padded to 96
#define UPD_K 128

typedef __attribute__((ext_vector_type(16))) __bf16 v16bf;
typedef __attribute__((ext_vector_type(8)))  float  v8f;
typedef __attribute__((ext_vector_type(4)))  int    v4i;

typedef __attribute__((address_space(1))) v4i* gv4i_p;
typedef __attribute__((address_space(3))) v4i* lv4i_p;

#if __has_builtin(__builtin_amdgcn_global_load_async_to_lds_b128)
#define HAS_ASYNC_LDS 1
#else
#define HAS_ASYNC_LDS 0
#endif

union BF16x16 { v16bf v; unsigned short u[16]; };

static __device__ __forceinline__ unsigned short f2bf(float f) {
    unsigned int x = __float_as_uint(f);
    unsigned int r = x + 0x7FFFu + ((x >> 16) & 1u);   // round-to-nearest-even
    return (unsigned short)(r >> 16);
}

static __device__ __forceinline__ void atomicMaxF(float* addr, float val) {
    unsigned int* a = (unsigned int*)addr;
    unsigned int old = *a;
    while (true) {
        float f = __uint_as_float(old);
        if (f >= val) break;
        unsigned int assumed = old;
        old = atomicCAS(a, assumed, __float_as_uint(val));
        if (old == assumed) break;
    }
}

// copy 16B global -> LDS (async path if available, else through VGPRs)
static __device__ __forceinline__ void g2l_b128(const void* g, void* l) {
#if HAS_ASYNC_LDS
    __builtin_amdgcn_global_load_async_to_lds_b128((gv4i_p)(v4i*)g, (lv4i_p)(v4i*)l, 0, 0);
#else
    *(uint4*)l = *(const uint4*)g;
#endif
}

static __device__ __forceinline__ void wait_stage() {
#if HAS_ASYNC_LDS
    asm volatile("s_wait_asynccnt 0x0" ::: "memory");
#endif
    asm volatile("s_wait_dscnt 0x0" ::: "memory");
}

// ---------------------------------------------------------------- prep weights
__global__ __launch_bounds__(256) void prep_weights(
    const float* __restrict__ Wmsg, const float* __restrict__ Wupd,
    unsigned short* __restrict__ wmsg_bf, unsigned short* __restrict__ wupd_bf) {
    int t = blockIdx.x * blockDim.x + threadIdx.x;
    const int nmsg = LL * MSG_K * HH;
    if (t < nmsg) {
        int l = t / (MSG_K * HH);
        int rem = t % (MSG_K * HH);
        int k = rem / HH, n = rem % HH;
        wmsg_bf[t] = (k < 88) ? f2bf(Wmsg[(l * 88 + k) * HH + n]) : (unsigned short)0;
    }
    int t2 = t - nmsg;
    if (t2 >= 0 && t2 < LL * UPD_K * HH) wupd_bf[t2] = f2bf(Wupd[t2]);
}

// ---------------------------------------------------------------- node embed
__global__ __launch_bounds__(256) void embed_kernel(
    const float* __restrict__ x, const float* __restrict__ W,
    const float* __restrict__ bv, const float* __restrict__ gln,
    const float* __restrict__ bln,
    float* __restrict__ h, unsigned short* __restrict__ hbf) {
    int wave = blockIdx.x * (blockDim.x >> 5) + (threadIdx.x >> 5);
    int lane = threadIdx.x & 31;
    if (wave >= NN) return;
    int i = wave;
    float a0 = bv[lane], a1 = bv[lane + 32];
    #pragma unroll
    for (int k = 0; k < 16; k++) {
        float xv = x[i * 16 + k];
        a0 += xv * W[k * HH + lane];
        a1 += xv * W[k * HH + lane + 32];
    }
    a0 = fmaxf(a0, 0.f); a1 = fmaxf(a1, 0.f);
    float s = a0 + a1, q = a0 * a0 + a1 * a1;
    #pragma unroll
    for (int off = 16; off >= 1; off >>= 1) {
        s += __shfl_xor(s, off, 32);
        q += __shfl_xor(q, off, 32);
    }
    float mu = s * (1.f / 64.f);
    float inv = rsqrtf(q * (1.f / 64.f) - mu * mu + 1e-5f);
    float o0 = (a0 - mu) * inv * gln[lane] + bln[lane];
    float o1 = (a1 - mu) * inv * gln[lane + 32] + bln[lane + 32];
    h[i * HH + lane] = o0;          h[i * HH + lane + 32] = o1;
    hbf[i * HH + lane] = f2bf(o0);  hbf[i * HH + lane + 32] = f2bf(o1);
}

// ---------------------------------------------------------------- edge messages (WMMA, double-buffered staging) + scatter
__global__ __launch_bounds__(256) void msg_kernel(
    const unsigned short* __restrict__ hbf,
    const int* __restrict__ ei,         // [2,E]: src then dst
    const float* __restrict__ eattr,    // [E,8]
    const int* __restrict__ egt,        // [E]
    const float* __restrict__ gemb,     // [NG,16] (layer slice)
    const unsigned short* __restrict__ Wb,  // [96,64] bf16
    const float* __restrict__ bias,     // [64]
    float* __restrict__ agg) {          // [N,64]
    __shared__ unsigned short sA[8][2][16 * MSG_K];   // 48 KB
    int w = threadIdx.x >> 5;
    int lane = threadIdx.x & 31;
    int colLocal = lane & 15;
    int khalf = lane >> 4;

    // B fragments: lane n holds column n%16, K-half selected by lane>=16
    BF16x16 B[3][4];
    #pragma unroll
    for (int kb = 0; kb < 3; kb++)
        #pragma unroll
        for (int nb = 0; nb < 4; nb++) {
            int col = nb * 16 + colLocal;
            int kbase = kb * 32 + khalf * 8;
            #pragma unroll
            for (int e = 0; e < 16; e++) B[kb][nb].u[e] = Wb[(kbase + (e < 8 ? e : e + 8)) * HH + col];
        }
    float bvv[4];
    #pragma unroll
    for (int nb = 0; nb < 4; nb++) bvv[nb] = bias[nb * 16 + colLocal];

    int r = lane >> 1;      // staged row (2 lanes per row)
    int half = lane & 1;

    auto stage = [&](int tt, int pb) {
        int e = tt * 16 + r;
        int srcn = ei[e];
        unsigned short* rowp = &sA[w][pb][r * MSG_K];
        const uint4* hp = (const uint4*)(hbf + (size_t)srcn * HH);
        #pragma unroll
        for (int qd = 0; qd < 4; qd++)                  // cols 0..63 (h), 4 chunks/lane
            g2l_b128(hp + half * 4 + qd, (uint4*)rowp + half * 4 + qd);
        if (half == 1) {                                // cols 64..95 need f32->bf16
            #pragma unroll
            for (int c = 0; c < 8; c++)  rowp[64 + c] = f2bf(eattr[e * 8 + c]);
            int gt = egt[e];
            #pragma unroll
            for (int c = 0; c < 16; c++) rowp[72 + c] = f2bf(gemb[gt * 16 + c]);
            #pragma unroll
            for (int c = 88; c < 96; c++) rowp[c] = 0;
        }
    };

    int wglobal = blockIdx.x * 8 + w;
    int nwaves = gridDim.x * 8;
    const int nt = EE / 16;
    int pb = 0;
    if (wglobal < nt) stage(wglobal, 0);
    for (int t = wglobal; t < nt; t += nwaves) {
        wait_stage();                                   // staging of buffer pb done
        const unsigned short* arow = &sA[w][pb][(lane & 15) * MSG_K];
        BF16x16 A[3];
        #pragma unroll
        for (int kb = 0; kb < 3; kb++) {
            int kbase = kb * 32 + khalf * 8;
            #pragma unroll
            for (int qe = 0; qe < 8; qe++) {
                A[kb].u[qe]     = arow[kbase + qe];
                A[kb].u[8 + qe] = arow[kbase + 16 + qe];
            }
        }
        int tn = t + nwaves;
        if (tn < nt) stage(tn, pb ^ 1);                 // overlap next gather with compute

        v8f acc0 = {}, acc1 = {}, acc2 = {}, acc3 = {};
        #pragma unroll
        for (int kb = 0; kb < 3; kb++) {
            acc0 = __builtin_amdgcn_wmma_f32_16x16x32_bf16(false, A[kb].v, false, B[kb][0].v, (short)0, acc0, false, false);
            acc1 = __builtin_amdgcn_wmma_f32_16x16x32_bf16(false, A[kb].v, false, B[kb][1].v, (short)0, acc1, false, false);
            acc2 = __builtin_amdgcn_wmma_f32_16x16x32_bf16(false, A[kb].v, false, B[kb][2].v, (short)0, acc2, false, false);
            acc3 = __builtin_amdgcn_wmma_f32_16x16x32_bf16(false, A[kb].v, false, B[kb][3].v, (short)0, acc3, false, false);
        }
        // scatter: lane holds col = nb*16+colLocal, rows v + khalf*8
        int dbase = EE + t * 16;
        #pragma unroll
        for (int v = 0; v < 8; v++) {
            int M = v + khalf * 8;
            int dn = ei[dbase + M];
            float* ap = agg + (size_t)dn * HH;
            atomicAdd(ap + ( 0 + colLocal), fmaxf(acc0[v] + bvv[0], 0.f));
            atomicAdd(ap + (16 + colLocal), fmaxf(acc1[v] + bvv[1], 0.f));
            atomicAdd(ap + (32 + colLocal), fmaxf(acc2[v] + bvv[2], 0.f));
            atomicAdd(ap + (48 + colLocal), fmaxf(acc3[v] + bvv[3], 0.f));
        }
        pb ^= 1;
    }
}

// ---------------------------------------------------------------- node update (WMMA) + residual
__global__ __launch_bounds__(256) void upd_kernel(
    const unsigned short* __restrict__ hbf_in,
    const float* __restrict__ agg,
    const unsigned short* __restrict__ Wb,   // [128,64] bf16
    const float* __restrict__ bias,
    float* __restrict__ h,
    unsigned short* __restrict__ hbf_out) {
    __shared__ unsigned short sA[8][16 * UPD_K];      // 32 KB
    int w = threadIdx.x >> 5;
    int lane = threadIdx.x & 31;
    int colLocal = lane & 15;
    int khalf = lane >> 4;

    BF16x16 B[4][4];
    #pragma unroll
    for (int kb = 0; kb < 4; kb++)
        #pragma unroll
        for (int nb = 0; nb < 4; nb++) {
            int col = nb * 16 + colLocal;
            int kbase = kb * 32 + khalf * 8;
            #pragma unroll
            for (int e = 0; e < 16; e++) B[kb][nb].u[e] = Wb[(kbase + (e < 8 ? e : e + 8)) * HH + col];
        }
    float bvv[4];
    #pragma unroll
    for (int nb = 0; nb < 4; nb++) bvv[nb] = bias[nb * 16 + colLocal];

    int r = lane >> 1;
    int half = lane & 1;
    unsigned short* rowp = &sA[w][r * UPD_K];
    const unsigned short* arow = &sA[w][(lane & 15) * UPD_K];

    int wglobal = blockIdx.x * 8 + w;
    int nwaves = gridDim.x * 8;
    for (int t = wglobal; t < NN / 16; t += nwaves) {
        int i0 = t * 16;
        int i = i0 + r;
        // cols 0..63 = h (16B copies, 4 chunks/lane)
        const uint4* hp = (const uint4*)(hbf_in + (size_t)i * HH);
        #pragma unroll
        for (int qd = 0; qd < 4; qd++)
            g2l_b128(hp + half * 4 + qd, (uint4*)rowp + half * 4 + qd);
        // cols 64..127 = bf16(agg), 32 values per lane
        const float4* gp = (const float4*)(agg + (size_t)i * HH + half * 32);
        #pragma unroll
        for (int qd = 0; qd < 8; qd++) {
            float4 v4 = gp[qd];
            unsigned short* op = rowp + 64 + half * 32 + qd * 4;
            op[0] = f2bf(v4.x); op[1] = f2bf(v4.y);
            op[2] = f2bf(v4.z); op[3] = f2bf(v4.w);
        }
        wait_stage();

        v8f acc0 = {}, acc1 = {}, acc2 = {}, acc3 = {};
        #pragma unroll
        for (int kb = 0; kb < 4; kb++) {
            BF16x16 A;
            int kbase = kb * 32 + khalf * 8;
            #pragma unroll
            for (int qe = 0; qe < 8; qe++) {
                A.u[qe]     = arow[kbase + qe];
                A.u[8 + qe] = arow[kbase + 16 + qe];
            }
            acc0 = __builtin_amdgcn_wmma_f32_16x16x32_bf16(false, A.v, false, B[kb][0].v, (short)0, acc0, false, false);
            acc1 = __builtin_amdgcn_wmma_f32_16x16x32_bf16(false, A.v, false, B[kb][1].v, (short)0, acc1, false, false);
            acc2 = __builtin_amdgcn_wmma_f32_16x16x32_bf16(false, A.v, false, B[kb][2].v, (short)0, acc2, false, false);
            acc3 = __builtin_amdgcn_wmma_f32_16x16x32_bf16(false, A.v, false, B[kb][3].v, (short)0, acc3, false, false);
        }
        #pragma unroll
        for (int v = 0; v < 8; v++) {
            int i2 = i0 + v + khalf * 8;
            size_t base = (size_t)i2 * HH;
            float n0 = h[base +  0 + colLocal] + acc0[v] + bvv[0];
            float n1 = h[base + 16 + colLocal] + acc1[v] + bvv[1];
            float n2 = h[base + 32 + colLocal] + acc2[v] + bvv[2];
            float n3 = h[base + 48 + colLocal] + acc3[v] + bvv[3];
            h[base +  0 + colLocal] = n0;  hbf_out[base +  0 + colLocal] = f2bf(n0);
            h[base + 16 + colLocal] = n1;  hbf_out[base + 16 + colLocal] = f2bf(n1);
            h[base + 32 + colLocal] = n2;  hbf_out[base + 32 + colLocal] = f2bf(n2);
            h[base + 48 + colLocal] = n3;  hbf_out[base + 48 + colLocal] = f2bf(n3);
        }
    }
}

// ---------------------------------------------------------------- pooling
__global__ __launch_bounds__(256) void pool_init(float* hsum, float* hmax, float* cnt) {
    int t = blockIdx.x * blockDim.x + threadIdx.x;
    if (t < BB * HH) { hsum[t] = 0.f; hmax[t] = -3.0e38f; }
    if (t < BB) cnt[t] = 0.f;
}

__global__ __launch_bounds__(256) void pool_kernel(
    const float* __restrict__ h, const int* __restrict__ batch,
    float* __restrict__ hsum, float* __restrict__ hmax, float* __restrict__ cnt) {
    long long t = (long long)blockIdx.x * blockDim.x + threadIdx.x;
    if (t >= (long long)NN * HH) return;
    int i = (int)(t >> 6), j = (int)(t & 63);
    int b = batch[i];
    float v = h[t];
    atomicAdd(&hsum[b * HH + j], v);
    atomicMaxF(&hmax[b * HH + j], v);
    if (j == 0) atomicAdd(&cnt[b], 1.f);
}

// ---------------------------------------------------------------- head (tiny MLP, per-graph block)
__global__ __launch_bounds__(128) void head_kernel(
    const float* __restrict__ gf,
    const float* __restrict__ Wg, const float* __restrict__ bg,
    const float* __restrict__ ggln, const float* __restrict__ bgln,
    const float* __restrict__ hsum, const float* __restrict__ hmax,
    const float* __restrict__ cnt,
    const float* __restrict__ Wc1, const float* __restrict__ bc1,
    const float* __restrict__ gcln, const float* __restrict__ bcln,
    const float* __restrict__ Wc2, const float* __restrict__ bc2,
    const float* __restrict__ Wh, const float* __restrict__ bh,
    float* __restrict__ out) {
    __shared__ float cvec[256];
    __shared__ float sc1[128];
    __shared__ float sc2[64];
    __shared__ float stats[2];
    int b = blockIdx.x, tid = threadIdx.x;

    if (tid < 64) {
        float a = bg[tid];
        for (int k = 0; k < 32; k++) a += gf[b * 32 + k] * Wg[k * 64 + tid];
        cvec[192 + tid] = fmaxf(a, 0.f);
        float c = cnt[b];
        float s = hsum[b * 64 + tid];
        cvec[tid] = s / fmaxf(c, 1.f);
        float m = hmax[b * 64 + tid];
        cvec[64 + tid] = (c > 0.f) ? m : 0.f;
        cvec[128 + tid] = s;
    }
    __syncthreads();
    if (tid == 0) {
        float s = 0.f, q = 0.f;
        for (int k = 0; k < 64; k++) { float v = cvec[192 + k]; s += v; q += v * v; }
        float mu = s / 64.f;
        stats[0] = mu; stats[1] = rsqrtf(q / 64.f - mu * mu + 1e-5f);
    }
    __syncthreads();
    if (tid < 64)
        cvec[192 + tid] = (cvec[192 + tid] - stats[0]) * stats[1] * ggln[tid] + bgln[tid];
    __syncthreads();
    {
        float a = bc1[tid];
        for (int k = 0; k < 256; k++) a += cvec[k] * Wc1[k * 128 + tid];
        sc1[tid] = fmaxf(a, 0.f);
    }
    __syncthreads();
    if (tid == 0) {
        float s = 0.f, q = 0.f;
        for (int k = 0; k < 128; k++) { float v = sc1[k]; s += v; q += v * v; }
        float mu = s / 128.f;
        stats[0] = mu; stats[1] = rsqrtf(q / 128.f - mu * mu + 1e-5f);
    }
    __syncthreads();
    float c1v = (sc1[tid] - stats[0]) * stats[1] * gcln[tid] + bcln[tid];
    __syncthreads();
    sc1[tid] = c1v;
    __syncthreads();
    if (tid < 64) {
        float a = bc2[tid];
        for (int k = 0; k < 128; k++) a += sc1[k] * Wc2[k * 64 + tid];
        sc2[tid] = fmaxf(a, 0.f);
    }
    __syncthreads();
    if (tid < CC) {
        float a = bh[tid];
        for (int k = 0; k < 64; k++) a += sc2[k] * Wh[k * CC + tid];
        out[b * CC + tid] = a;
    }
}

extern "C" void kernel_launch(void* const* d_in, const int* in_sizes, int n_in,
                              void* d_out, int out_size, void* d_ws, size_t ws_size,
                              hipStream_t stream) {
    (void)in_sizes; (void)n_in; (void)out_size; (void)ws_size;
    const float* x       = (const float*)d_in[0];
    const int*   ei      = (const int*)d_in[1];
    const float* eattr   = (const float*)d_in[2];
    const int*   egt     = (const int*)d_in[3];
    const int*   batch   = (const int*)d_in[4];
    const float* gf      = (const float*)d_in[5];
    const float* W_embed = (const float*)d_in[6];
    const float* b_embed = (const float*)d_in[7];
    const float* g_eln   = (const float*)d_in[8];
    const float* b_eln   = (const float*)d_in[9];
    const float* gemb    = (const float*)d_in[10];   // [L,NG,16]
    const float* Wmsg    = (const float*)d_in[11];   // [L,88,64]
    const float* bmsg    = (const float*)d_in[12];   // [L,64]
    const float* Wupd    = (const float*)d_in[13];   // [L,128,64]
    const float* bupd    = (const float*)d_in[14];   // [L,64]
    const float* Wgproj  = (const float*)d_in[15];
    const float* bgproj  = (const float*)d_in[16];
    const float* ggln    = (const float*)d_in[17];
    const float* bgln    = (const float*)d_in[18];
    const float* Wc1     = (const float*)d_in[19];
    const float* bc1     = (const float*)d_in[20];
    const float* gcln    = (const float*)d_in[21];
    const float* bcln    = (const float*)d_in[22];
    const float* Wc2     = (const float*)d_in[23];
    const float* bc2     = (const float*)d_in[24];
    const float* Whead   = (const float*)d_in[25];
    const float* bhead   = (const float*)d_in[26];

    char* ws = (char*)d_ws;
    float* h              = (float*)(ws + 0);                       // 12,800,000 B
    float* agg            = (float*)(ws + 12800000);                // 12,800,000 B
    unsigned short* hbf   = (unsigned short*)(ws + 25600000);       //  6,400,000 B
    unsigned short* wmsgb = (unsigned short*)(ws + 32000000);       //     49,152 B
    unsigned short* wupdb = (unsigned short*)(ws + 32049152);       //     65,536 B
    float* hsum           = (float*)(ws + 32114688);                //     32,768 B
    float* hmax           = (float*)(ws + 32147456);                //     32,768 B
    float* cntb           = (float*)(ws + 32180224);                //        512 B

    prep_weights<<<224, 256, 0, stream>>>(Wmsg, Wupd, wmsgb, wupdb);
    embed_kernel<<<(NN + 7) / 8, 256, 0, stream>>>(x, W_embed, b_embed, g_eln, b_eln, h, hbf);

    for (int l = 0; l < LL; l++) {
        (void)hipMemsetAsync(agg, 0, (size_t)NN * HH * sizeof(float), stream);
        msg_kernel<<<640, 256, 0, stream>>>(hbf, ei, eattr, egt,
                                            gemb + (size_t)l * NGT * GEW,
                                            wmsgb + (size_t)l * MSG_K * HH,
                                            bmsg + l * HH, agg);
        upd_kernel<<<391, 256, 0, stream>>>(hbf, agg,
                                            wupdb + (size_t)l * UPD_K * HH,
                                            bupd + l * HH, h, hbf);
    }

    pool_init<<<32, 256, 0, stream>>>(hsum, hmax, cntb);
    pool_kernel<<<(NN * HH + 255) / 256, 256, 0, stream>>>(h, batch, hsum, hmax, cntb);
    head_kernel<<<BB, 128, 0, stream>>>(gf, Wgproj, bgproj, ggln, bgln,
                                        hsum, hmax, cntb,
                                        Wc1, bc1, gcln, bcln, Wc2, bc2,
                                        Whead, bhead, (float*)d_out);
}
